// SeqGenRNN_72086731096553
// MI455X (gfx1250) — compile-verified, matrix operand order checked by reading, not run on hardware
//
#include <hip/hip_runtime.h>
#include <stdint.h>

// ---------------------------------------------------------------------------
// CDNA5 (gfx1250) seq2seq RNN: bf16 WMMA everywhere.
//   V=32000 E=H=512 L=2 B=32 T=64
// ---------------------------------------------------------------------------

typedef __attribute__((ext_vector_type(16))) __bf16       v16bf;
typedef __attribute__((ext_vector_type(8)))  float        v8f;
typedef __attribute__((ext_vector_type(4)))  unsigned int u32x4;

union ABChunk { u32x4 q[2]; v16bf v; };

// ---- async global->LDS staging (CDNA5 ASYNCcnt path), with safe fallback ---
#if __has_builtin(__builtin_amdgcn_global_load_async_to_lds_b128)
#define HAVE_ASYNC_LDS 1
typedef __attribute__((ext_vector_type(4))) int v4i;
typedef __attribute__((address_space(1))) v4i as1_v4i;   // global int4
typedef __attribute__((address_space(3))) v4i as3_v4i;   // LDS int4
#endif

__device__ __forceinline__ void lds_copy16(u32x4* l, const uint16_t* g) {
#ifdef HAVE_ASYNC_LDS
  __builtin_amdgcn_global_load_async_to_lds_b128((as1_v4i*)g, (as3_v4i*)l, 0, 0);
#else
  *l = *(const u32x4*)g;
#endif
}

__device__ __forceinline__ void async_join() {
#ifdef HAVE_ASYNC_LDS
#if __has_builtin(__builtin_amdgcn_s_wait_asynccnt)
  __builtin_amdgcn_s_wait_asynccnt(0);
#else
  asm volatile("s_wait_asynccnt 0x0" ::: "memory");
#endif
#endif
}

__device__ __forceinline__ uint16_t f2bf(float f) {
  union { float f; uint32_t u; } x; x.f = f;
  uint32_t u = x.u;
  return (uint16_t)((u + 0x7FFFu + ((u >> 16) & 1u)) >> 16);  // RNE
}

// B-tile (K=32 x N=16, bf16) intra-tile mapping, mirror of the ISA 16-bit
// A layout (CDNA5 ISA 7.12.2): lane%16 = column N, lane/16 picks
// the K half; element e (0..15) -> k_local.
__device__ __forceinline__ void bmap(int lane, int e, int& kl, int& nl) {
  int half = lane >> 4;
  nl = lane & 15;
  int v = e >> 1;
  int base = (v < 4) ? 0 : 16;
  kl = base + 2 * (v & 3) + (e & 1) + 8 * half;
}

// ---------------------------------------------------------------------------
// Pack fc_W [32000,512] f32 -> bf16 B-tile blocks [2000 nt][16 kc][32 lane][16 e]
// ---------------------------------------------------------------------------
__global__ void pack_fc_kernel(const float* __restrict__ W,
                               uint16_t* __restrict__ dst) {
  int idx = blockIdx.x * 256 + threadIdx.x;
  if (idx >= 2000 * 16 * 512) return;
  int e    = idx & 15;
  int lane = (idx >> 4) & 31;
  int kc   = (idx >> 9) & 15;
  int nt   = idx >> 13;
  int kl, nl; bmap(lane, e, kl, nl);
  int k = kc * 32 + kl;
  int n = nt * 16 + nl;
  dst[idx] = f2bf(W[(size_t)n * 512 + k]);
}

// ---------------------------------------------------------------------------
// Pack RNN weights: Wcat[k,n] = k<512 ? Wih[l][n][k] : Whh[l][n][k-512]
// dst layout: [rl(4)][32 nt][32 kc][32 lane][16 e]   (rl = rnn*2+layer)
// ---------------------------------------------------------------------------
__global__ void pack_rnn_kernel(const float* __restrict__ eWi, const float* __restrict__ eWh,
                                const float* __restrict__ dWi, const float* __restrict__ dWh,
                                uint16_t* __restrict__ dst) {
  int idx = blockIdx.x * 256 + threadIdx.x;
  if (idx >= 4 * 32 * 32 * 512) return;
  int e    = idx & 15;
  int lane = (idx >> 4) & 31;
  int kc   = (idx >> 9) & 31;
  int nt   = (idx >> 14) & 31;
  int rl   = idx >> 19;
  int l = rl & 1;
  const float* Wi = (rl >> 1) ? dWi : eWi;
  const float* Wh = (rl >> 1) ? dWh : eWh;
  int kl, nl; bmap(lane, e, kl, nl);
  int k = kc * 32 + kl;            // 0..1023
  int n = nt * 16 + nl;            // 0..511
  float v = (k < 512) ? Wi[((size_t)l * 512 + n) * 512 + k]
                      : Wh[((size_t)l * 512 + n) * 512 + (k - 512)];
  dst[idx] = f2bf(v);
}

// bias sums: dst [rnn(2)][layer(2)][512] = bih + bhh
__global__ void pack_bias_kernel(const float* __restrict__ ebi, const float* __restrict__ ebh,
                                 const float* __restrict__ dbi, const float* __restrict__ dbh,
                                 float* __restrict__ dst) {
  int idx = blockIdx.x * 256 + threadIdx.x;
  if (idx >= 2048) return;
  int n = idx & 511;
  int l = (idx >> 9) & 1;
  int r = idx >> 10;
  const float* bi = r ? dbi : ebi;
  const float* bh = r ? dbh : ebh;
  dst[idx] = bi[l * 512 + n] + bh[l * 512 + n];
}

// embedding gather: out[t][b][k] = bf16(emb[tok[b*64+t]][k]),  t in [0,T)
__global__ void emb_kernel(const float* __restrict__ emb, const int* __restrict__ tok,
                           uint16_t* __restrict__ out, int T) {
  int idx = blockIdx.x * 256 + threadIdx.x;
  if (idx >= T * 32 * 512) return;
  int k = idx & 511;
  int b = (idx >> 9) & 31;
  int t = idx >> 14;
  int token = tok[b * 64 + t];
  out[idx] = f2bf(emb[(size_t)token * 512 + k]);
}

// ---------------------------------------------------------------------------
// Persistent 2-layer RNN. 16 WGs x 128 threads (4 waves). Wave -> one 16x16
// output tile of h_t. Weights for this WG's 2 N-tiles live in LDS (64 KB),
// staged with async global->LDS loads. Per step:
//   C[32,512] = Xcat[32,1024] @ Wcat[1024,512]; h = tanh(C + b).
// Grid barrier (global atomic + s_sleep spin) between steps; hidden state is
// double-buffered bf16 so writers can't race readers across the barrier.
// ---------------------------------------------------------------------------
__global__ void rnn_kernel(const uint16_t* __restrict__ x0,    // [T][32][512] layer-0 input
                           uint16_t* __restrict__ tmp,         // [T][32][512] layer-0 output
                           uint16_t* __restrict__ out1,        // [T][32][512] layer-1 output
                           const uint16_t* __restrict__ wpk,   // [2 layers][32nt][32kc][512]
                           const float* __restrict__ bias,     // [2][512] (bih+bhh)
                           uint16_t* __restrict__ hst,         // [2 parity][2 layer][32][512]
                           unsigned int* __restrict__ cnt,
                           int T) {
  extern __shared__ u32x4 smem[];                 // 4096 * 16B = 64 KB
  const int tid   = threadIdx.x;
  const int wg    = blockIdx.x;                   // 0..15
  const int wave  = tid >> 5;                     // 0..3
  const int lane  = tid & 31;
  const int mtile = wave >> 1;                    // 0..1  (M = 32)
  const int ntl   = wave & 1;                     // local N-tile
  const int ntile = wg * 2 + ntl;                 // 0..31 (N = 512)
  const int half  = lane >> 4;
  const int mrow  = mtile * 16 + (lane & 15);     // A row this lane loads
  unsigned int bar = 0;

  for (int l = 0; l < 2; ++l) {
    // stage this layer's weight slice (2 N-tiles * K=1024) into LDS
    const uint16_t* wsrc = wpk + (size_t)l * 524288 + (size_t)wg * 32768;
    for (int i = tid; i < 4096; i += 128) lds_copy16(smem + i, wsrc + (size_t)i * 8);
    async_join();
    __syncthreads();

    const uint16_t* xin  = (l == 0) ? x0 : tmp;
    uint16_t*       sout = (l == 0) ? tmp : out1;
    const float*    bl   = bias + l * 512;

    for (int t = 0; t < T; ++t) {
      const uint16_t* hrd = hst + (size_t)(((t & 1) * 2 + l)) * 32 * 512;
      uint16_t*       hwr = hst + (size_t)((((t + 1) & 1) * 2 + l)) * 32 * 512;
      const uint16_t* xrow = xin + ((size_t)t * 32 + mrow) * 512;
      const uint16_t* hrow = hrd + (size_t)mrow * 512;

      v8f acc = {};
      for (int c = 0; c < 32; ++c) {              // K = 1024 -> 32 wmma
        ABChunk A, B;
        const uint16_t* src = (c < 16) ? (xrow + 32 * c) : (hrow + 32 * (c - 16));
        A.q[0] = *(const u32x4*)(src + 8 * half);
        A.q[1] = *(const u32x4*)(src + 16 + 8 * half);
        int bo = ((ntl * 32 + c) * 32 + lane) * 2;
        B.q[0] = smem[bo];
        B.q[1] = smem[bo + 1];
        acc = __builtin_amdgcn_wmma_f32_16x16x32_bf16(false, A.v, false, B.v,
                                                      (short)0, acc, false, false);
      }

      // epilogue: h = tanh(acc + bias); write bf16 to state + sequence buf
      int n = ntile * 16 + (lane & 15);
      float bv = bl[n];
#pragma unroll
      for (int r = 0; r < 8; ++r) {
        int m = mtile * 16 + r + 8 * half;
        uint16_t hb = f2bf(tanhf(acc[r] + bv));
        hwr[(size_t)m * 512 + n] = hb;
        sout[((size_t)t * 32 + m) * 512 + n] = hb;
      }

      // grid-wide barrier (monotonic counter, goal = bar * nWG)
      __threadfence();
      __syncthreads();
      ++bar;
      if (tid == 0) {
        atomicAdd(cnt, 1u);
        while (__hip_atomic_load(cnt, __ATOMIC_RELAXED, __HIP_MEMORY_SCOPE_AGENT)
               < bar * gridDim.x)
          __builtin_amdgcn_s_sleep(1);
      }
      __syncthreads();
      __threadfence();
    }
    __syncthreads();
  }
}

// ---------------------------------------------------------------------------
// logits[2016,32000] = dec_seq @ fc_W^T + fc_b.
// Block = 256 threads / 8 waves. Each block stages a 16-N-tile x K=512 weight
// slice (256 KB) into LDS ONCE (async), then sweeps 18 M-tiles against it.
// Each wave owns 2 N-tiles; A (16x32 bf16 chunks) streams from global.
// B traffic: 7 x 32 MB (L2-resident) instead of 126 x 32 MB.
// Output row m = b*63+t maps 1:1 to the [B,63,V] layout.
// ---------------------------------------------------------------------------
__global__ void fc_kernel(const uint16_t* __restrict__ aseq,  // [63][32][512] bf16
                          const uint16_t* __restrict__ wpk,   // packed fc_W tiles
                          const float* __restrict__ bias,     // [32000]
                          float* __restrict__ out) {          // [2016][32000]
  extern __shared__ u32x4 smem[];                  // 16 nt * 16 KB = 256 KB
  const int tid  = threadIdx.x;
  const int wave = tid >> 5;
  const int lane = tid & 31;
  const int half = lane >> 4;
  const int ngrp = blockIdx.x;                     // 0..124 (16 N-tiles each)
  const int mgrp = blockIdx.y;                     // 0..6   (18 M-tiles each)

  // stage the B slice for this block's 16 N-tiles (contiguous 256 KB)
  const uint16_t* bsrc = wpk + (size_t)ngrp * 131072;
  for (int i = tid; i < 16384; i += 256) lds_copy16(smem + i, bsrc + (size_t)i * 8);
  async_join();
  __syncthreads();

  const int ntl = wave * 2;                        // this wave's 2 local N-tiles
  const int bo1 = 1024;                            // u32x4 stride between N-tiles

  for (int mi = 0; mi < 18; ++mi) {
    const int mtile = mgrp * 18 + mi;
    const int mrow  = mtile * 16 + (lane & 15);
    const int b = mrow / 63, t = mrow % 63;
    const uint16_t* arow = aseq + ((size_t)t * 32 + b) * 512;
    if (mi + 1 < 18) {                             // global_prefetch_b8 next A row
      int mr2 = (mtile + 1) * 16 + (lane & 15);
      __builtin_prefetch((const void*)(aseq + ((size_t)(mr2 % 63) * 32 + mr2 / 63) * 512), 0, 1);
    }

    v8f a0 = {}, a1 = {};
    for (int c = 0; c < 16; ++c) {                 // K = 512
      ABChunk A, B0, B1;
      A.q[0] = *(const u32x4*)(arow + 32 * c + 8 * half);
      A.q[1] = *(const u32x4*)(arow + 32 * c + 16 + 8 * half);
      int bo = ((ntl * 16 + c) * 32 + lane) * 2;
      B0.q[0] = smem[bo];       B0.q[1] = smem[bo + 1];
      B1.q[0] = smem[bo + bo1]; B1.q[1] = smem[bo + bo1 + 1];
      a0 = __builtin_amdgcn_wmma_f32_16x16x32_bf16(false, A.v, false, B0.v,
                                                   (short)0, a0, false, false);
      a1 = __builtin_amdgcn_wmma_f32_16x16x32_bf16(false, A.v, false, B1.v,
                                                   (short)0, a1, false, false);
    }

    const int n0 = (ngrp * 16 + ntl) * 16 + (lane & 15);
    const int n1 = n0 + 16;
    const float bv0 = bias[n0];
    const float bv1 = bias[n1];
#pragma unroll
    for (int r = 0; r < 8; ++r) {
      int m = mtile * 16 + r + 8 * half;           // == b*63 + t of that row
      out[(size_t)m * 32000 + n0] = a0[r] + bv0;
      out[(size_t)m * 32000 + n1] = a1[r] + bv1;
    }
  }
}

// ---------------------------------------------------------------------------
extern "C" void kernel_launch(void* const* d_in, const int* in_sizes, int n_in,
                              void* d_out, int out_size, void* d_ws, size_t ws_size,
                              hipStream_t stream) {
  const int*   src     = (const int*)  d_in[0];
  const int*   trg     = (const int*)  d_in[1];
  const float* enc_emb = (const float*)d_in[2];
  const float* dec_emb = (const float*)d_in[3];
  const float* eWi     = (const float*)d_in[4];
  const float* eWh     = (const float*)d_in[5];
  const float* ebi     = (const float*)d_in[6];
  const float* ebh     = (const float*)d_in[7];
  const float* dWi     = (const float*)d_in[8];
  const float* dWh     = (const float*)d_in[9];
  const float* dbi     = (const float*)d_in[10];
  const float* dbh     = (const float*)d_in[11];
  const float* fcW     = (const float*)d_in[12];
  const float* fcb     = (const float*)d_in[13];

  char* ws = (char*)d_ws;
  size_t off = 0;
  auto alloc = [&](size_t bytes) {
    char* p = ws + off;
    off += (bytes + 255) & ~(size_t)255;
    return p;
  };
  uint16_t* FCW  = (uint16_t*)alloc(2000u * 16 * 512 * 2);   // 32.77 MB
  uint16_t* RNNW = (uint16_t*)alloc(4u * 32 * 32 * 512 * 2); //  4.19 MB
  float*    BSUM = (float*)   alloc(2048 * 4);
  uint16_t* XBE  = (uint16_t*)alloc(64u * 32 * 512 * 2);     // enc embeddings
  uint16_t* XBD  = (uint16_t*)alloc(64u * 32 * 512 * 2);     // dec embeddings
  uint16_t* TMPA = (uint16_t*)alloc(64u * 32 * 512 * 2);     // layer-0 outputs
  uint16_t* TMPB = (uint16_t*)alloc(64u * 32 * 512 * 2);     // layer-1 outputs
  uint16_t* HST  = (uint16_t*)alloc(2u * 2 * 32 * 512 * 2);  // double-buffered h
  unsigned* CNT  = (unsigned*)alloc(256);                    // 2 barrier counters
  (void)ws_size; (void)in_sizes; (void)n_in; (void)out_size;

  // zero hidden state + barrier counters (HST and CNT are adjacent)
  (void)hipMemsetAsync((void*)HST, 0, 2u * 2 * 32 * 512 * 2 + 256, stream);

  pack_fc_kernel  <<<(2000 * 16 * 512 + 255) / 256, 256, 0, stream>>>(fcW, FCW);
  pack_rnn_kernel <<<(4 * 32 * 32 * 512 + 255) / 256, 256, 0, stream>>>(eWi, eWh, dWi, dWh, RNNW);
  pack_bias_kernel<<<8, 256, 0, stream>>>(ebi, ebh, dbi, dbh, BSUM);
  emb_kernel      <<<(64 * 32 * 512 + 255) / 256, 256, 0, stream>>>(enc_emb, src, XBE, 64);
  emb_kernel      <<<(63 * 32 * 512 + 255) / 256, 256, 0, stream>>>(dec_emb, trg, XBD, 63);

  // encoder: h starts at 0, finals land in HST parity 0 (T=64 even)
  rnn_kernel<<<16, 128, 65536, stream>>>(XBE, TMPA, TMPB, RNNW, BSUM, HST, CNT, 64);
  // decoder: continues from encoder finals in HST; layer-1 output -> TMPB
  rnn_kernel<<<16, 128, 65536, stream>>>(XBD, TMPA, TMPB, RNNW + 2u * 524288,
                                         BSUM + 1024, HST, CNT + 1, 63);
  // vocab projection: 125 N-groups x 7 M-groups, 256 KB LDS each
  fc_kernel<<<dim3(125, 7), 256, 262144, stream>>>(TMPB, FCW, fcb, (float*)d_out);
}
